// NRU_v1_50766513439530
// MI455X (gfx1250) — matrix operand order repaired and checked by reading, compiled, not run on hardware
//
#include <hip/hip_runtime.h>
#include <math.h>

#define BB 4
#define LL 8192
#define DM 512            // D_MODEL == D_MEM
#define KAV 512
#define NAV 2048
#define SSM 1024

typedef __bf16 bf16;
typedef __attribute__((ext_vector_type(16))) __bf16 bfrag16;
typedef __attribute__((ext_vector_type(8)))  __bf16 bfrag8;
typedef __attribute__((ext_vector_type(8)))  float  v8f;
typedef __attribute__((ext_vector_type(4)))  unsigned u32x4;
typedef __attribute__((ext_vector_type(8)))  unsigned u32x8;

__device__ __forceinline__ bf16 f2bf(float f) {
  unsigned u = __builtin_bit_cast(unsigned, f);
  unsigned r = u + 0x7fffu + ((u >> 16) & 1u);   // round-to-nearest-even
  unsigned short h = (unsigned short)(r >> 16);
  return __builtin_bit_cast(bf16, h);
}

// A-matrix 16x32 bf16 fragment (ISA 7.12.2): lane<16 holds K {0..7,16..23},
// lane>=16 holds K {8..15,24..31} of row (lane&15). Two contiguous 16B loads.
__device__ __forceinline__ bfrag16 load_a_frag(const bf16* p, int half) {
  union { bfrag16 v; bfrag8 h[2]; } f;
  f.h[0] = *(const bfrag8*)(p + half * 8);
  f.h[1] = *(const bfrag8*)(p + 16 + half * 8);
  return f.v;
}

// B-matrix 32x16 bf16 fragment: lane (n + 16*half) holds K {8*half..+7, 16+8*half..+7} of
// column n = W row n. Two contiguous 16B loads from the row-major weight row.
__device__ __forceinline__ bfrag16 load_b_frag(const bf16* wrow_k0, int half) {
  union { bfrag16 v; bfrag8 h[2]; } f;
  f.h[0] = *(const bfrag8*)(wrow_k0 + half * 8);
  f.h[1] = *(const bfrag8*)(wrow_k0 + 16 + half * 8);
  return f.v;
}

#define WMMA_BF16(A, B, C) \
  __builtin_amdgcn_wmma_f32_16x16x32_bf16(false, (A), false, (B), (short)0, (C), false, false)

// ----------------------------------------------------------- TDM tile loader
// Async-load a 16-row x 512-col bf16 tile (row stride 512 elems) from global
// into LDS via the Tensor Data Mover (CDNA5 ISA-8).
// D# group0: count=1, lds_addr, 57b global tile addr, type=2 ("image").
// D# group1: data_size=1 (2B), tensor_dim0=512, tensor_dim1=16, tile=512x16,
//            tensor_dim0_stride=512. Groups 2/3 zero (2D tile, dims 3/4 unused).
__device__ __forceinline__ void tdm_load_tile_16x512(const bf16* gsrc, void* lds_dst) {
  unsigned long long ga = (unsigned long long)(uintptr_t)gsrc;
  unsigned lds_off = (unsigned)(uintptr_t)lds_dst;       // LDS byte offset
  u32x4 g0;
  g0[0] = 1u;                                            // count=1, user desc
  g0[1] = lds_off;                                       // lds_addr [63:32]
  g0[2] = (unsigned)ga;                                  // global_addr lo
  g0[3] = (unsigned)((ga >> 32) & 0x01FFFFFFu) | (2u << 30);  // addr hi | type=2
  u32x8 g1;
  g1[0] = (1u << 16);                                    // data_size=1 (2 bytes)
  g1[1] = (512u & 0xFFFFu) << 16;                        // tensor_dim0 lo16 @ [63:48]
  g1[2] = (512u >> 16) | ((16u & 0xFFFFu) << 16);        // tensor_dim0 hi | tensor_dim1 lo
  g1[3] = (16u >> 16) | (512u << 16);                    // tensor_dim1 hi | tile_dim0=512
  g1[4] = 16u;                                           // tile_dim1=16, tile_dim2=0
  g1[5] = 512u;                                          // tensor_dim0_stride lo32
  g1[6] = 0u;                                            // stride hi | dim1_stride lo
  g1[7] = 0u;
  u32x4 gz = {0u, 0u, 0u, 0u};
  asm volatile("tensor_load_to_lds %0, %1, %2, %3"
               :: "s"(g0), "s"(g1), "s"(gz), "s"(gz)
               : "memory");
}

// ---------------------------------------------------------------- conversions
__global__ void nru_cvt_bf16(const float* __restrict__ in, bf16* __restrict__ out, int n) {
  int i = blockIdx.x * blockDim.x + threadIdx.x;
  int stride = gridDim.x * blockDim.x;
  for (; i < n; i += stride) out[i] = f2bf(in[i]);
}

__global__ void nru_zero_row0(float* __restrict__ msf) {
  int i = blockIdx.x * blockDim.x + threadIdx.x;      // 2048 threads
  int b = i >> 9, d = i & 511;
  msf[(size_t)b * LL * DM + d] = 0.f;                 // ms[b, 0, :] = 0
}

// ------------------------------------------------- GEMM1 fused with NRU combine
// Block: 8 waves share one (b, mt) A tile staged in LDS by TDM; each wave owns
// one d_mem column tile and computes the 4 chunk tiles sharing the A fragment:
// ms_inp = (av0+b0)*(av1+b1) - (av2+b2)*(av3+b3), written to ms[b, l+1, d].
__global__ void nru_gemm_av(const bf16* __restrict__ xb, const bf16* __restrict__ wav,
                            const float* __restrict__ bav, float* __restrict__ msf) {
  __shared__ bf16 lds_a[16 * KAV];                     // 16 KB A tile
  const int lane = threadIdx.x & 31;
  const int wave = threadIdx.x >> 5;
  const int half = lane >> 4;
  const int nl   = lane & 15;
  const int nt = blockIdx.x * 8 + wave;                // 0..31 column tile
  const int mt = blockIdx.y;                           // 0..511 row tile
  const int b  = blockIdx.z;

  if (wave == 0) {                                     // TDM: EXEC-independent
    tdm_load_tile_16x512(xb + ((size_t)b * LL + mt * 16) * DM, lds_a);
    __builtin_amdgcn_s_wait_tensorcnt(0);
  }
  __syncthreads();

  const bf16* aptr = lds_a + (size_t)nl * KAV;         // A row nl from LDS
  const bf16* w0 = wav + (size_t)(0 * DM + nt * 16 + nl) * KAV;
  const bf16* w1 = wav + (size_t)(1 * DM + nt * 16 + nl) * KAV;
  const bf16* w2 = wav + (size_t)(2 * DM + nt * 16 + nl) * KAV;
  const bf16* w3 = wav + (size_t)(3 * DM + nt * 16 + nl) * KAV;

  v8f c0 = {}, c1 = {}, c2 = {}, c3 = {};
#pragma unroll 4
  for (int k0 = 0; k0 < KAV; k0 += 32) {
    bfrag16 a = load_a_frag(aptr + k0, half);
    c0 = WMMA_BF16(a, load_b_frag(w0 + k0, half), c0);
    c1 = WMMA_BF16(a, load_b_frag(w1 + k0, half), c1);
    c2 = WMMA_BF16(a, load_b_frag(w2 + k0, half), c2);
    c3 = WMMA_BF16(a, load_b_frag(w3 + k0, half), c3);
  }
  const int col = nt * 16 + nl;
  const float ba0 = bav[col], ba1 = bav[DM + col];
  const float ba2 = bav[2 * DM + col], ba3 = bav[3 * DM + col];
#pragma unroll
  for (int r = 0; r < 8; ++r) {
    int m = mt * 16 + r + 8 * half;                    // C layout: row r+8*half
    if (m <= LL - 2) {                                 // only 8191 valid rows
      float v = (c0[r] + ba0) * (c1[r] + ba1) - (c2[r] + ba2) * (c3[r] + ba3);
      msf[((size_t)b * LL + m + 1) * DM + col] = v;
    }
  }
}

// ---------------------------------------------------------------- chunked scan
__global__ void nru_scan_partial(const float* __restrict__ msf, float* __restrict__ cs) {
  int idx = blockIdx.x * blockDim.x + threadIdx.x;    // 4*64*512 = 131072
  int d = idx & 511, c = (idx >> 9) & 63, b = idx >> 15;
  const float* p = msf + ((size_t)b * LL + c * 128) * DM + d;
  float s = 0.f;
  for (int i = 0; i < 128; ++i) s += p[(size_t)i * DM];
  cs[idx] = s;
}

__global__ void nru_scan_chunks(float* __restrict__ cs) {
  int idx = blockIdx.x * blockDim.x + threadIdx.x;    // 2048
  int d = idx & 511, b = idx >> 9;
  float run = 0.f;
  for (int c = 0; c < 64; ++c) {
    size_t o = ((size_t)b * 64 + c) * DM + d;
    float t = cs[o]; cs[o] = run; run += t;           // exclusive scan in place
  }
}

__global__ void nru_scan_apply(const float* __restrict__ msf, const float* __restrict__ cs,
                               bf16* __restrict__ msb) {
  int idx = blockIdx.x * blockDim.x + threadIdx.x;    // 131072
  int d = idx & 511, c = (idx >> 9) & 63, b = idx >> 15;
  float run = cs[idx];
  const float* p = msf + ((size_t)b * LL + c * 128) * DM + d;
  bf16* q = msb + ((size_t)b * LL + c * 128) * DM + d;
  for (int i = 0; i < 128; ++i) {
    run += p[(size_t)i * DM];
    q[(size_t)i * DM] = f2bf(run);                    // inclusive scan, bf16 out
  }
}

// ------------------------------------- GEMM2 (+concat fused into K) + exact GELU
// Block = one mt; waves = 8 N-groups of 4 tiles. Two TDM loads stage the x-half
// and ms-half of the A tile (K split of the channel concat) into LDS.
__global__ void nru_gemm_go(const bf16* __restrict__ xb, const bf16* __restrict__ msb,
                            const bf16* __restrict__ wgo, const float* __restrict__ bgo,
                            bf16* __restrict__ yb) {
  __shared__ bf16 lds_a[2 * 16 * DM];                 // 32 KB: x tile | ms tile
  const int lane = threadIdx.x & 31;
  const int wave = threadIdx.x >> 5;
  const int half = lane >> 4;
  const int nl   = lane & 15;
  const int mt = blockIdx.x;                          // 0..2047
  const int ng = wave;                                // 4 column tiles each

  if (wave == 0) {
    tdm_load_tile_16x512(xb  + (size_t)mt * 16 * DM, lds_a);
    tdm_load_tile_16x512(msb + (size_t)mt * 16 * DM, lds_a + 16 * DM);
    __builtin_amdgcn_s_wait_tensorcnt(0);
  }
  __syncthreads();

  const bf16* ax = lds_a + (size_t)nl * DM;
  const bf16* am = lds_a + 16 * DM + (size_t)nl * DM;

  const bf16* w[4];
#pragma unroll
  for (int t = 0; t < 4; ++t)
    w[t] = wgo + (size_t)((ng * 4 + t) * 16 + nl) * SSM;

  v8f acc[4] = {};
#pragma unroll 2
  for (int k0 = 0; k0 < DM; k0 += 32) {               // K 0..511 from x
    bfrag16 a = load_a_frag(ax + k0, half);
#pragma unroll
    for (int t = 0; t < 4; ++t)
      acc[t] = WMMA_BF16(a, load_b_frag(w[t] + k0, half), acc[t]);
  }
#pragma unroll 2
  for (int k0 = 0; k0 < DM; k0 += 32) {               // K 512..1023 from ms
    bfrag16 a = load_a_frag(am + k0, half);
#pragma unroll
    for (int t = 0; t < 4; ++t)
      acc[t] = WMMA_BF16(a, load_b_frag(w[t] + DM + k0, half), acc[t]);
  }
#pragma unroll
  for (int t = 0; t < 4; ++t) {
    int col = (ng * 4 + t) * 16 + nl;
    float bb = bgo[col];
#pragma unroll
    for (int r = 0; r < 8; ++r) {
      size_t row = (size_t)mt * 16 + r + 8 * half;
      float v = acc[t][r] + bb;
      float g = 0.5f * v * (1.f + erff(v * 0.70710678118654752f));   // exact gelu
      yb[row * DM + col] = f2bf(g);
    }
  }
}

// --------------------------------------------- GEMM3 with GLU fused (paired tiles)
// Block = one mt (blockIdx.x>>2); waves own paired column tiles (n, n+512).
__global__ void nru_gemm_out(const bf16* __restrict__ yb, const bf16* __restrict__ wout,
                             const float* __restrict__ bout, float* __restrict__ out) {
  __shared__ bf16 lds_a[16 * DM];                     // 16 KB A tile
  const int lane = threadIdx.x & 31;
  const int wave = threadIdx.x >> 5;
  const int half = lane >> 4;
  const int nl   = lane & 15;
  const int mt = blockIdx.x >> 2;                     // 0..2047
  const int pt = (blockIdx.x & 3) * 8 + wave;         // paired tile (n, n+512)

  if (wave == 0) {
    tdm_load_tile_16x512(yb + (size_t)mt * 16 * DM, lds_a);
    __builtin_amdgcn_s_wait_tensorcnt(0);
  }
  __syncthreads();

  const bf16* ap = lds_a + (size_t)nl * DM;
  const bf16* wg  = wout + (size_t)(pt * 16 + nl) * DM;        // z[:, n]
  const bf16* wsg = wout + (size_t)(DM + pt * 16 + nl) * DM;   // z[:, n+512]

  v8f cg = {}, cs2 = {};
#pragma unroll 4
  for (int k0 = 0; k0 < DM; k0 += 32) {
    bfrag16 a = load_a_frag(ap + k0, half);
    cg  = WMMA_BF16(a, load_b_frag(wg  + k0, half), cg);
    cs2 = WMMA_BF16(a, load_b_frag(wsg + k0, half), cs2);
  }
  int col = pt * 16 + nl;
  float bg = bout[col], bs = bout[DM + col];
#pragma unroll
  for (int r = 0; r < 8; ++r) {
    size_t row = (size_t)mt * 16 + r + 8 * half;
    float zg = cg[r] + bg;
    float zs = cs2[r] + bs;
    out[row * DM + col] = zg * (1.f / (1.f + expf(-zs)));     // GLU
  }
}

// ----------------------------------------------------------------------- launch
extern "C" void kernel_launch(void* const* d_in, const int* in_sizes, int n_in,
                              void* d_out, int out_size, void* d_ws, size_t ws_size,
                              hipStream_t stream) {
  (void)in_sizes; (void)n_in; (void)out_size; (void)ws_size;
  const float* x     = (const float*)d_in[0];
  const float* W_av  = (const float*)d_in[1];
  const float* b_av  = (const float*)d_in[2];
  const float* W_go  = (const float*)d_in[3];
  const float* b_go  = (const float*)d_in[4];
  const float* W_out = (const float*)d_in[5];
  const float* b_out = (const float*)d_in[6];
  float* out = (float*)d_out;

  char* ws = (char*)d_ws;
  size_t off = 0;
  auto alloc = [&](size_t bytes) -> char* {
    char* p = ws + off;
    off = (off + bytes + 255) & ~(size_t)255;
    return p;
  };
  bf16*  xb    = (bf16*) alloc((size_t)BB * LL * DM * 2);   // x in bf16
  bf16*  wavb  = (bf16*) alloc((size_t)NAV * KAV * 2);
  bf16*  wgob  = (bf16*) alloc((size_t)DM * SSM * 2);
  bf16*  woutb = (bf16*) alloc((size_t)SSM * DM * 2);
  bf16*  msb   = (bf16*) alloc((size_t)BB * LL * DM * 2);   // scanned memory, bf16
  bf16*  y_b   = (bf16*) alloc((size_t)BB * LL * DM * 2);   // gelu output, bf16
  float* msf   = (float*)alloc((size_t)BB * LL * DM * 4);   // ms_inp / pre-scan f32
  float* cs    = (float*)alloc((size_t)BB * 64 * DM * 4);   // chunk sums

  nru_cvt_bf16<<<4096, 256, 0, stream>>>(x, xb, BB * LL * DM);
  nru_cvt_bf16<<<1024, 256, 0, stream>>>(W_av, wavb, NAV * KAV);
  nru_cvt_bf16<<<512, 256, 0, stream>>>(W_go, wgob, DM * SSM);
  nru_cvt_bf16<<<512, 256, 0, stream>>>(W_out, woutb, SSM * DM);
  nru_zero_row0<<<8, 256, 0, stream>>>(msf);

  nru_gemm_av<<<dim3(4, 512, BB), 256, 0, stream>>>(xb, wavb, b_av, msf);

  nru_scan_partial<<<512, 256, 0, stream>>>(msf, cs);
  nru_scan_chunks<<<8, 256, 0, stream>>>(cs);
  nru_scan_apply<<<512, 256, 0, stream>>>(msf, cs, msb);

  nru_gemm_go<<<2048, 256, 0, stream>>>(xb, msb, wgob, b_go, y_b);
  nru_gemm_out<<<8192, 256, 0, stream>>>(y_b, woutb, b_out, out);
}
